// MultiHeadSparseMMAttentionSBH_52381421142525
// MI455X (gfx1250) — compile-verified, matrix operand order checked by reading, not run on hardware
//
#include <hip/hip_runtime.h>
#include <hip/hip_bf16.h>

// ---------------------------------------------------------------------------
// MultiHeadSparseMMAttentionSBH for MI455X (gfx1250), wave32 + WMMA f16.
// ---------------------------------------------------------------------------

typedef __attribute__((ext_vector_type(8)))  _Float16 v8h;
typedef __attribute__((ext_vector_type(16))) _Float16 v16h;
typedef __attribute__((ext_vector_type(8)))  float    v8f;

#define NUM_HEADS 16
#define HEAD_DIM  64
#define INNER     1024
#define SPARSE_N  8
#define S_VIS     8192
#define T_ENC     256
#define G_SEQ     1024          // S_VIS / SPARSE_N
#define L_TOT     1280          // G_SEQ + T_ENC
#define EPS_RMS   1e-5f

// ---- WMMA helpers ---------------------------------------------------------

__device__ __forceinline__ v8f wmma16(v16h a, v16h b, v8f c) {
    // D = A(16x32 f16) x B(32x16 f16) + C(16x16 f32)
    return __builtin_amdgcn_wmma_f32_16x16x32_f16(
        false, a, false, b, (short)0, c, false, false);
}

// Load one 16x32 f16 fragment (A layout; B assumed mirrored with lane=N).
// Source: 16 "rows" (M for A / N for B), each contiguous along K with
// leading dimension `ld` (in halves). Per ISA layout:
//   lanes 0-15 : row=lane, elems 0..7 = K 0..7,  elems 8..15 = K 16..23
//   lanes 16-31: row=lane-16, elems 0..7 = K 8..15, elems 8..15 = K 24..31
__device__ __forceinline__ v16h load_frag(const _Float16* base, int ld) {
    int lane = threadIdx.x & 31;
    int row  = lane & 15;
    int half = lane >> 4;
    const _Float16* p = base + (size_t)row * ld + half * 8;
    v8h lo = *(const v8h*)(p);
    v8h hi = *(const v8h*)(p + 16);
    return __builtin_shufflevector(lo, hi, 0,1,2,3,4,5,6,7,8,9,10,11,12,13,14,15);
}

__device__ __forceinline__ float red_max16(float x) {
    #pragma unroll
    for (int m = 1; m < 16; m <<= 1) x = fmaxf(x, __shfl_xor(x, m, 32));
    return x;
}
__device__ __forceinline__ float red_sum16(float x) {
    #pragma unroll
    for (int m = 1; m < 16; m <<= 1) x += __shfl_xor(x, m, 32);
    return x;
}

// ---- f32 -> f16 conversion ------------------------------------------------

__global__ void cvt_f16_kernel(const float* __restrict__ in,
                               _Float16* __restrict__ out, int n) {
    int i = blockIdx.x * blockDim.x + threadIdx.x;
    if (i < n) out[i] = (_Float16)in[i];
}

// ---- Projection: out = x @ W^T, per-head RMSNorm (+RoPE), scatter to
//      attention layout. One wave = 16 tokens x 1 head (64 cols). ----------

__global__ void __launch_bounds__(128)
proj_kernel(const _Float16* __restrict__ xh,   // [ntok][1024] f16
            const _Float16* __restrict__ wh,   // [1024][1024] f16 (row e, contiguous d)
            const float* __restrict__ g,       // [64] rms gamma, or nullptr
            const float* __restrict__ rcos,    // [S][64] rope cos, or nullptr
            const float* __restrict__ rsin,    // [S][64]
            int enc,                           // 0 = visual, 1 = encoder (broadcast)
            int transposed,                    // 0: [bh][pos][64], 1: [bh][64][1280]
            float qscale,
            _Float16* __restrict__ outbuf) {
    const int wave = threadIdx.x >> 5;
    const int lane = threadIdx.x & 31;
    const int id   = blockIdx.x * (blockDim.x >> 5) + wave;
    const int h    = id & 15;
    const int tile = id >> 4;
    const int s0   = tile * 16;
    const int n    = lane & 15;
    const int half = lane >> 4;

    v8f c[4] = {v8f{0}, v8f{0}, v8f{0}, v8f{0}};
    const _Float16* abase = xh + (size_t)s0 * INNER;
    const _Float16* bbase = wh + (size_t)(h * HEAD_DIM) * INNER;

    for (int kk = 0; kk < INNER; kk += 32) {
        v16h a = load_frag(abase + kk, INNER);
        #pragma unroll
        for (int t = 0; t < 4; ++t) {
            v16h b = load_frag(bbase + (size_t)(t * 16) * INNER + kk, INNER);
            c[t] = wmma16(a, b, c[t]);
        }
    }

    // Per-head RMS norm over the 64 columns of this head.
    if (g != nullptr) {
        float rinv[8];
        #pragma unroll
        for (int r = 0; r < 8; ++r) {
            float ss = 0.f;
            #pragma unroll
            for (int t = 0; t < 4; ++t) ss += c[t][r] * c[t][r];
            ss = red_sum16(ss);
            rinv[r] = rsqrtf(ss * (1.0f / HEAD_DIM) + EPS_RMS);
        }
        #pragma unroll
        for (int t = 0; t < 4; ++t) {
            float gv = g[t * 16 + n];
            #pragma unroll
            for (int r = 0; r < 8; ++r) c[t][r] *= rinv[r] * gv;
        }
    }

    // RoPE (visual q/k only). rot = concat(-x2, x1): tiles 0,1 pair with 2,3.
    if (rcos != nullptr) {
        #pragma unroll
        for (int r = 0; r < 8; ++r) {
            int s = s0 + r + half * 8;
            float o0 = c[0][r], o1 = c[1][r], o2 = c[2][r], o3 = c[3][r];
            float cv0 = rcos[(size_t)s * 64 +  0 + n], sv0 = rsin[(size_t)s * 64 +  0 + n];
            float cv1 = rcos[(size_t)s * 64 + 16 + n], sv1 = rsin[(size_t)s * 64 + 16 + n];
            float cv2 = rcos[(size_t)s * 64 + 32 + n], sv2 = rsin[(size_t)s * 64 + 32 + n];
            float cv3 = rcos[(size_t)s * 64 + 48 + n], sv3 = rsin[(size_t)s * 64 + 48 + n];
            c[0][r] = o0 * cv0 - o2 * sv0;
            c[1][r] = o1 * cv1 - o3 * sv1;
            c[2][r] = o2 * cv2 + o0 * sv2;
            c[3][r] = o3 * cv3 + o1 * sv3;
        }
    }

    if (qscale != 1.0f) {
        #pragma unroll
        for (int t = 0; t < 4; ++t)
            #pragma unroll
            for (int r = 0; r < 8; ++r) c[t][r] *= qscale;
    }

    // Scatter into attention layout.
    #pragma unroll
    for (int r = 0; r < 8; ++r) {
        int row = r + half * 8;
        int tok = s0 + row;
        #pragma unroll
        for (int t = 0; t < 4; ++t) {
            _Float16 hv = (_Float16)c[t][r];
            int d = t * 16 + n;
            if (!enc) {
                int j = tok & (SPARSE_N - 1);
                int pos = tok >> 3;
                size_t bh = (size_t)(j * NUM_HEADS + h);
                size_t off = transposed
                    ? (bh * HEAD_DIM + d) * L_TOT + pos
                    : (bh * L_TOT + pos) * HEAD_DIM + d;
                outbuf[off] = hv;
            } else {
                int pos = G_SEQ + tok;
                #pragma unroll
                for (int j = 0; j < SPARSE_N; ++j) {
                    size_t bh = (size_t)(j * NUM_HEADS + h);
                    size_t off = transposed
                        ? (bh * HEAD_DIM + d) * L_TOT + pos
                        : (bh * L_TOT + pos) * HEAD_DIM + d;
                    outbuf[off] = hv;
                }
            }
        }
    }
}

// ---- Flash attention per (batch j, head h): L=1280, d=64 ------------------

__global__ void __launch_bounds__(128)
attn_kernel(const _Float16* __restrict__ qbuf,  // [128][1280][64] (Q pre-scaled)
            const _Float16* __restrict__ kbuf,  // [128][1280][64]
            const _Float16* __restrict__ vbuf,  // [128][64][1280] (transposed)
            _Float16* __restrict__ avis,        // [8192][1024]
            float* __restrict__ otxt) {         // [256][1024], zero-initialized
    const int wave = threadIdx.x >> 5;
    const int lane = threadIdx.x & 31;
    const int bh   = blockIdx.x;            // j*16 + h
    const int qt   = blockIdx.y * 4 + wave; // 0..79
    const int h    = bh & 15;
    const int j    = bh >> 4;
    const int n    = lane & 15;
    const int half = lane >> 4;
    const int q0   = qt * 16;

    const _Float16* Q  = qbuf + (size_t)bh * L_TOT * HEAD_DIM;
    const _Float16* K  = kbuf + (size_t)bh * L_TOT * HEAD_DIM;
    const _Float16* VT = vbuf + (size_t)bh * HEAD_DIM * L_TOT;

    __shared__ __align__(16) _Float16 pst[4][16 * 32];
    _Float16* pw = pst[wave];

    v16h qa0 = load_frag(Q + (size_t)q0 * HEAD_DIM +  0, HEAD_DIM);
    v16h qa1 = load_frag(Q + (size_t)q0 * HEAD_DIM + 32, HEAD_DIM);

    v8f acc[4] = {v8f{0}, v8f{0}, v8f{0}, v8f{0}};
    float m_i[8], l_i[8];
    #pragma unroll
    for (int r = 0; r < 8; ++r) { m_i[r] = -1e30f; l_i[r] = 0.f; }

    for (int k0 = 0; k0 < L_TOT; k0 += 32) {
        // ---- scores: two 16x16 tiles, d chained 0..31 / 32..63
        v16h kb0 = load_frag(K + (size_t)(k0)      * HEAD_DIM +  0, HEAD_DIM);
        v16h kb1 = load_frag(K + (size_t)(k0)      * HEAD_DIM + 32, HEAD_DIM);
        v16h kb2 = load_frag(K + (size_t)(k0 + 16) * HEAD_DIM +  0, HEAD_DIM);
        v16h kb3 = load_frag(K + (size_t)(k0 + 16) * HEAD_DIM + 32, HEAD_DIM);
        v8f sc0 = v8f{0}; sc0 = wmma16(qa0, kb0, sc0); sc0 = wmma16(qa1, kb1, sc0);
        v8f sc1 = v8f{0}; sc1 = wmma16(qa0, kb2, sc1); sc1 = wmma16(qa1, kb3, sc1);

        // ---- online softmax update (per-row stats live in r-slots)
        float p0[8], p1[8], alph[8];
        #pragma unroll
        for (int r = 0; r < 8; ++r) {
            float mx = red_max16(fmaxf(sc0[r], sc1[r]));
            float mn = fmaxf(m_i[r], mx);
            alph[r] = __expf(m_i[r] - mn);
            m_i[r] = mn;
            p0[r] = __expf(sc0[r] - mn);
            p1[r] = __expf(sc1[r] - mn);
            l_i[r] = l_i[r] * alph[r] + red_sum16(p0[r] + p1[r]);
        }
        #pragma unroll
        for (int t = 0; t < 4; ++t)
            #pragma unroll
            for (int r = 0; r < 8; ++r) acc[t][r] *= alph[r];

        // ---- restage P (D layout -> A fragment) through LDS
        #pragma unroll
        for (int r = 0; r < 8; ++r) {
            int row = r + half * 8;
            pw[row * 32 + n]      = (_Float16)p0[r];
            pw[row * 32 + 16 + n] = (_Float16)p1[r];
        }
        asm volatile("s_wait_dscnt 0x0" ::: "memory");
        v16h pa = load_frag(pw, 32);

        // ---- acc += P(16x32) x V(32x64)
        #pragma unroll
        for (int t = 0; t < 4; ++t) {
            v16h vb = load_frag(VT + (size_t)(t * 16) * L_TOT + k0, L_TOT);
            acc[t] = wmma16(pa, vb, acc[t]);
        }
    }

    // ---- epilogue: normalize and scatter
    #pragma unroll
    for (int r = 0; r < 8; ++r) {
        int row = r + half * 8;
        int pos = q0 + row;
        float inv = 1.0f / l_i[r];
        #pragma unroll
        for (int t = 0; t < 4; ++t) {
            float val = acc[t][r] * inv;
            int e = h * HEAD_DIM + t * 16 + n;
            if (pos < G_SEQ) {
                int s = pos * SPARSE_N + j;
                avis[(size_t)s * INNER + e] = (_Float16)val;
            } else {
                atomicAdd(&otxt[(size_t)(pos - G_SEQ) * INNER + e],
                          val * (1.0f / SPARSE_N));
            }
        }
    }
}

// ---- Output GEMM: out = A @ W^T + bias (f32 out) --------------------------

__global__ void __launch_bounds__(128)
out_gemm_kernel(const _Float16* __restrict__ A,   // [M][1024]
                const _Float16* __restrict__ wh,  // [1024][1024]
                const float* __restrict__ bias,   // [1024]
                float* __restrict__ out) {        // [M][1024]
    const int wave = threadIdx.x >> 5;
    const int lane = threadIdx.x & 31;
    const int id   = blockIdx.x * (blockDim.x >> 5) + wave;
    const int cg   = id & 15;
    const int tile = id >> 4;
    const int s0   = tile * 16;
    const int c0   = cg * 64;
    const int n    = lane & 15;
    const int half = lane >> 4;

    v8f c[4] = {v8f{0}, v8f{0}, v8f{0}, v8f{0}};
    const _Float16* abase = A + (size_t)s0 * INNER;
    const _Float16* bbase = wh + (size_t)c0 * INNER;

    for (int kk = 0; kk < INNER; kk += 32) {
        v16h a = load_frag(abase + kk, INNER);
        #pragma unroll
        for (int t = 0; t < 4; ++t) {
            v16h b = load_frag(bbase + (size_t)(t * 16) * INNER + kk, INNER);
            c[t] = wmma16(a, b, c[t]);
        }
    }

    #pragma unroll
    for (int r = 0; r < 8; ++r) {
        int row = r + half * 8;
        #pragma unroll
        for (int t = 0; t < 4; ++t) {
            int e = c0 + t * 16 + n;
            out[(size_t)(s0 + row) * INNER + e] = c[t][r] + bias[e];
        }
    }
}

// ---------------------------------------------------------------------------

extern "C" void kernel_launch(void* const* d_in, const int* in_sizes, int n_in,
                              void* d_out, int out_size, void* d_ws, size_t ws_size,
                              hipStream_t stream) {
    (void)in_sizes; (void)n_in; (void)out_size; (void)ws_size;

    const float* hs   = (const float*)d_in[0];   // [8192][1][1024]
    const float* ehs  = (const float*)d_in[1];   // [256][1][1024]
    const float* Wq   = (const float*)d_in[2];
    const float* Wk   = (const float*)d_in[3];
    const float* Wv   = (const float*)d_in[4];
    const float* Waq  = (const float*)d_in[5];
    const float* Wak  = (const float*)d_in[6];
    const float* Wav  = (const float*)d_in[7];
    const float* Wout = (const float*)d_in[8];
    const float* bout = (const float*)d_in[9];
    const float* Waout= (const float*)d_in[10];
    const float* baout= (const float*)d_in[11];
    const float* gq   = (const float*)d_in[12];
    const float* gk   = (const float*)d_in[13];
    const float* gaq  = (const float*)d_in[14];
    const float* gak  = (const float*)d_in[15];
    const float* rc   = (const float*)d_in[16];  // [8192][64]
    const float* rs   = (const float*)d_in[17];

    float* out = (float*)d_out;                  // [8448][1024]

    // ---- workspace carve-up
    char* ws = (char*)d_ws;
    size_t off = 0;
    auto carve = [&](size_t bytes) -> char* {
        char* p = ws + off;
        off += (bytes + 255) & ~(size_t)255;
        return p;
    };
    _Float16* xh_vis = (_Float16*)carve((size_t)S_VIS * INNER * 2);
    _Float16* xh_enc = (_Float16*)carve((size_t)T_ENC * INNER * 2);
    _Float16* whq    = (_Float16*)carve((size_t)INNER * INNER * 2);
    _Float16* whk    = (_Float16*)carve((size_t)INNER * INNER * 2);
    _Float16* whv    = (_Float16*)carve((size_t)INNER * INNER * 2);
    _Float16* whaq   = (_Float16*)carve((size_t)INNER * INNER * 2);
    _Float16* whak   = (_Float16*)carve((size_t)INNER * INNER * 2);
    _Float16* whav   = (_Float16*)carve((size_t)INNER * INNER * 2);
    _Float16* whout  = (_Float16*)carve((size_t)INNER * INNER * 2);
    _Float16* whaout = (_Float16*)carve((size_t)INNER * INNER * 2);
    _Float16* qbuf   = (_Float16*)carve((size_t)128 * L_TOT * HEAD_DIM * 2);
    _Float16* kbuf   = (_Float16*)carve((size_t)128 * L_TOT * HEAD_DIM * 2);
    _Float16* vbuf   = (_Float16*)carve((size_t)128 * L_TOT * HEAD_DIM * 2);
    _Float16* avis   = (_Float16*)carve((size_t)S_VIS * INNER * 2);
    float*    otxt   = (float*)   carve((size_t)T_ENC * INNER * 4);
    _Float16* atxt   = (_Float16*)carve((size_t)T_ENC * INNER * 2);

    // ---- stage 0: f32 -> f16 conversions
    auto cvt = [&](const float* src, _Float16* dst, int nelem) {
        cvt_f16_kernel<<<(nelem + 255) / 256, 256, 0, stream>>>(src, dst, nelem);
    };
    cvt(hs,    xh_vis, S_VIS * INNER);
    cvt(ehs,   xh_enc, T_ENC * INNER);
    cvt(Wq,    whq,    INNER * INNER);
    cvt(Wk,    whk,    INNER * INNER);
    cvt(Wv,    whv,    INNER * INNER);
    cvt(Waq,   whaq,   INNER * INNER);
    cvt(Wak,   whak,   INNER * INNER);
    cvt(Wav,   whav,   INNER * INNER);
    cvt(Wout,  whout,  INNER * INNER);
    cvt(Waout, whaout, INNER * INNER);

    hipMemsetAsync(otxt, 0, (size_t)T_ENC * INNER * 4, stream);

    // ---- stage 1: projections (+RMS/RoPE) into attention layout
    const float qscale = 0.125f;  // 1/sqrt(HEAD_DIM), folded into Q
    // visual: 512 token tiles * 16 heads = 8192 waves -> 2048 blocks of 4 waves
    proj_kernel<<<2048, 128, 0, stream>>>(xh_vis, whq, gq, rc, rs, 0, 0, qscale, qbuf);
    proj_kernel<<<2048, 128, 0, stream>>>(xh_vis, whk, gk, rc, rs, 0, 0, 1.0f,   kbuf);
    proj_kernel<<<2048, 128, 0, stream>>>(xh_vis, whv, nullptr, nullptr, nullptr,
                                          0, 1, 1.0f, vbuf);
    // encoder: 16 token tiles * 16 heads = 256 waves -> 64 blocks
    proj_kernel<<<64, 128, 0, stream>>>(xh_enc, whaq, gaq, nullptr, nullptr,
                                        1, 0, qscale, qbuf);
    proj_kernel<<<64, 128, 0, stream>>>(xh_enc, whak, gak, nullptr, nullptr,
                                        1, 0, 1.0f, kbuf);
    proj_kernel<<<64, 128, 0, stream>>>(xh_enc, whav, nullptr, nullptr, nullptr,
                                        1, 1, 1.0f, vbuf);

    // ---- stage 2: attention. 128 (j,h) pairs x 80 query tiles (4 waves/block)
    attn_kernel<<<dim3(128, 20), 128, 0, stream>>>(qbuf, kbuf, vbuf, avis, otxt);

    // ---- stage 3: text mean -> f16
    cvt(otxt, atxt, T_ENC * INNER);

    // ---- stage 4: output projections
    out_gemm_kernel<<<2048, 128, 0, stream>>>(avis, whout, bout, out);
    out_gemm_kernel<<<64,   128, 0, stream>>>(atxt, whaout, baout,
                                              out + (size_t)S_VIS * INNER);
}
